// HSMSSD_5085241279223
// MI455X (gfx1250) — matrix operand order repaired and checked by reading, not compiled, not run on hardware
//
#include <hip/hip_runtime.h>

typedef __attribute__((ext_vector_type(16))) _Float16 v16h;
typedef __attribute__((ext_vector_type(8)))  _Float16 v8h;
typedef __attribute__((ext_vector_type(4)))  _Float16 v4h;
typedef __attribute__((ext_vector_type(8)))  float    v8f;

// Load 4 contiguous elements as f16x4 (converting if the source is f32).
__device__ __forceinline__ v4h ld4h(const float* p) {
    const float4 v = *(const float4*)p;
    v4h h; h[0] = (_Float16)v.x; h[1] = (_Float16)v.y;
           h[2] = (_Float16)v.z; h[3] = (_Float16)v.w;
    return h;
}
__device__ __forceinline__ v4h ld4h(const _Float16* p) { return *(const v4h*)p; }

// ---------------------------------------------------------------------------
// Tiled WMMA GEMM: C[b] = A[b](MxK) * B[b](KxN), f16 math, f32 accumulate/out.
// Block = 128 threads (4 waves). Block tile 64(M) x 64(N), K stepped by 32.
// Each wave owns a 16-wide N strip and sweeps 4 stacked 16x16 WMMA tiles,
// reusing one B fragment across 4 A fragments (4 back-to-back v_wmma).
// Requires M%64==0, N%64==0, K%32==0 (true for every GEMM here).
// ---------------------------------------------------------------------------
template <typename TA, typename TB>
__global__ __launch_bounds__(128)
void gemm_wmma(const TA* __restrict__ A, long long sA, int lda,
               const TB* __restrict__ B, long long sB, int ldb,
               float* __restrict__ C, long long sC, int ldc, int K)
{
    __shared__ _Float16 As[64][32];   // A tile, row-major [m][k]
    __shared__ _Float16 Bts[64][32];  // B tile transposed [n][k]

    const int m0 = blockIdx.y * 64;
    const int n0 = blockIdx.x * 64;
    const TA* Ab = A + (long long)blockIdx.z * sA;
    const TB* Bb = B + (long long)blockIdx.z * sB;
    float*    Cb = C + (long long)blockIdx.z * sC;

    const int t     = threadIdx.x;
    const int wave  = t >> 5;
    const int lane  = t & 31;
    const int nloc  = lane & 15;         // col (B,D) / row (A) within 16
    const int g     = (lane >> 4) & 1;   // lane half-group
    const int nbase = wave * 16;         // this wave's N strip

    // Cooperative-load coordinates: 4 chunks of 4 elements for A and B.
    // A tile: 64 rows x 8 chunks; B tile: 32 rows x 16 chunks.
    int aRow[4], aCol[4], bRow[4], bCol[4];
    #pragma unroll
    for (int i = 0; i < 4; ++i) {
        const int f = t + 128 * i;
        aRow[i] = f >> 3;   aCol[i] = (f & 7) * 4;
        bRow[i] = f >> 4;   bCol[i] = (f & 15) * 4;
    }

    v8f acc[4] = {{}, {}, {}, {}};

    for (int k0 = 0; k0 < K; k0 += 32) {
        // ---- phase 1: vector loads (+convert) into registers --------------
        v4h ar[4], br[4];
        #pragma unroll
        for (int i = 0; i < 4; ++i)
            ar[i] = ld4h(&Ab[(m0 + aRow[i]) * lda + (k0 + aCol[i])]);
        #pragma unroll
        for (int i = 0; i < 4; ++i)
            br[i] = ld4h(&Bb[(k0 + bRow[i]) * ldb + (n0 + bCol[i])]);

        if (k0 + 32 < K) {  // L2 prefetch of next tiles
            __builtin_prefetch(&Ab[(m0 + aRow[0]) * lda + (k0 + 32 + aCol[0])], 0, 1);
            __builtin_prefetch(&Bb[(k0 + 32 + bRow[0]) * ldb + (n0 + bCol[0])], 0, 1);
        }

        __syncthreads();  // previous iteration's fragment reads are done

        // ---- phase 2: stage to LDS ----------------------------------------
        #pragma unroll
        for (int i = 0; i < 4; ++i)
            *(v4h*)&As[aRow[i]][aCol[i]] = ar[i];        // ds_store_b64
        #pragma unroll
        for (int i = 0; i < 4; ++i) {                    // transposed scatter
            Bts[bCol[i] + 0][bRow[i]] = br[i][0];
            Bts[bCol[i] + 1][bRow[i]] = br[i][1];
            Bts[bCol[i] + 2][bRow[i]] = br[i][2];
            Bts[bCol[i] + 3][bRow[i]] = br[i][3];
        }
        __syncthreads();

        // ---- phase 3: fragments + 4x WMMA ---------------------------------
        // B fragment (32x16): lane(n,g) elem e -> K=16g+e, N=n
        v8h blo = *(const v8h*)&Bts[nbase + nloc][16 * g];
        v8h bhi = *(const v8h*)&Bts[nbase + nloc][16 * g + 8];
        v16h bv;
        #pragma unroll
        for (int i = 0; i < 8; ++i) { bv[i] = blo[i]; bv[i + 8] = bhi[i]; }

        #pragma unroll
        for (int mt = 0; mt < 4; ++mt) {
            // A fragment (16x32): lane(m,g) elem e -> K=16g+e, M=m
            v8h alo = *(const v8h*)&As[16 * mt + nloc][16 * g];
            v8h ahi = *(const v8h*)&As[16 * mt + nloc][16 * g + 8];
            v16h av;
            #pragma unroll
            for (int i = 0; i < 8; ++i) { av[i] = alo[i]; av[i + 8] = ahi[i]; }
            acc[mt] = __builtin_amdgcn_wmma_f32_16x16x32_f16(
                false, av, false, bv, (short)0, acc[mt], false, false);
        }
    }

    // D layout: VGPR r -> row (16*mt + r + 8g), col = lane%16
    #pragma unroll
    for (int mt = 0; mt < 4; ++mt)
        #pragma unroll
        for (int r = 0; r < 8; ++r)
            Cb[(m0 + 16 * mt + r + 8 * g) * ldc + (n0 + nbase + nloc)] = acc[mt][r];
}

// ---------------------------------------------------------------------------
// One-shot f32 -> f16 conversion (vectorized), used for x.
// ---------------------------------------------------------------------------
__global__ __launch_bounds__(256)
void cvt_f32_f16(const float* __restrict__ in, _Float16* __restrict__ out,
                 long long n4)
{
    const long long i = (long long)blockIdx.x * 256 + threadIdx.x;
    if (i >= n4) return;
    *(v4h*)(out + i * 4) = ld4h(in + i * 4);
}

// ---------------------------------------------------------------------------
// Depthwise 3x3 SAME conv: one (batch, channel) 64x64 image per block.
// Channels 64..127 (the Cm operand of the final GEMM) are also emitted as f16.
// ---------------------------------------------------------------------------
__global__ __launch_bounds__(256)
void dwconv3x3(const float* __restrict__ in, const float* __restrict__ Wdw,
               float* __restrict__ out, _Float16* __restrict__ outh)
{
    __shared__ float tile[64 * 64];
    const int c = blockIdx.x % 192;
    const int b = blockIdx.x / 192;
    const long long base = ((long long)b * 192 + c) * 4096;
    const int t = threadIdx.x;

    float w9[9];
    #pragma unroll
    for (int j = 0; j < 9; ++j) w9[j] = Wdw[c * 9 + j];

    #pragma unroll
    for (int i = 0; i < 16; ++i) tile[t + 256 * i] = in[base + t + 256 * i];
    __syncthreads();

    const bool isCm = (c >= 64) && (c < 128);
    _Float16* oh = outh + ((long long)b * 64 + (c - 64)) * 4096;

    #pragma unroll
    for (int i = 0; i < 16; ++i) {
        int idx = t + 256 * i;
        int h = idx >> 6, w = idx & 63;
        float s = 0.f;
        #pragma unroll
        for (int dh = 0; dh < 3; ++dh) {
            int hh = h + dh - 1;
            if (hh < 0 || hh > 63) continue;
            #pragma unroll
            for (int dw2 = 0; dw2 < 3; ++dw2) {
                int ww = w + dw2 - 1;
                if (ww < 0 || ww > 63) continue;
                s += tile[hh * 64 + ww] * w9[dh * 3 + dw2];
            }
        }
        out[base + idx] = s;
        if (isCm) oh[idx] = (_Float16)s;
    }
}

// ---------------------------------------------------------------------------
// Row softmax over L=4096 of (dt + A_param[s]) then AB = A * Bm;
// writes AB transposed AND in f16: abt[b][l][s], the B operand of G2.
// ---------------------------------------------------------------------------
__global__ __launch_bounds__(256)
void softmax_ab(const float* __restrict__ dw, const float* __restrict__ A_param,
                _Float16* __restrict__ abt)
{
    __shared__ float red[256];
    const int s = blockIdx.x & 63;
    const int b = blockIdx.x >> 6;
    const float Ap = A_param[s];
    const float* dtrow = dw + ((long long)b * 192 + 128 + s) * 4096;
    const float* brow  = dw + ((long long)b * 192 + s) * 4096;
    _Float16* abtb = abt + (long long)b * 4096 * 64;
    const int t = threadIdx.x;

    float lmax = -3.4e38f;
    for (int l = t; l < 4096; l += 256) lmax = fmaxf(lmax, dtrow[l] + Ap);
    red[t] = lmax; __syncthreads();
    for (int st = 128; st > 0; st >>= 1) {
        if (t < st) red[t] = fmaxf(red[t], red[t + st]);
        __syncthreads();
    }
    const float rmax = red[0]; __syncthreads();

    float lsum = 0.f;
    for (int l = t; l < 4096; l += 256) lsum += __expf(dtrow[l] + Ap - rmax);
    red[t] = lsum; __syncthreads();
    for (int st = 128; st > 0; st >>= 1) {
        if (t < st) red[t] += red[t + st];
        __syncthreads();
    }
    const float inv = 1.f / red[0];

    for (int l = t; l < 4096; l += 256) {
        float av = __expf(dtrow[l] + Ap - rmax) * inv;
        abtb[(long long)l * 64 + s] = (_Float16)(av * brow[l]);
    }
}

// ---------------------------------------------------------------------------
// hgate = h1 * silu(z) + h1 * D    (h1 = hz[:,0:384,:], z = hz[:,384:768,:])
// ---------------------------------------------------------------------------
__global__ __launch_bounds__(256)
void gate_silu(const float* __restrict__ hz, const float* __restrict__ Dp,
               float* __restrict__ hg)
{
    long long i = (long long)blockIdx.x * 256 + threadIdx.x;
    if (i >= 16LL * 384 * 64) return;
    const long long bi  = i / (384 * 64);
    const long long rem = i % (384 * 64);
    const float h1 = hz[bi * 768 * 64 + rem];
    const float z  = hz[bi * 768 * 64 + 384LL * 64 + rem];
    const float sig = 1.f / (1.f + __expf(-z));
    hg[i] = h1 * (z * sig) + h1 * Dp[0];
}

// ---------------------------------------------------------------------------
extern "C" void kernel_launch(void* const* d_in, const int* in_sizes, int n_in,
                              void* d_out, int out_size, void* d_ws, size_t ws_size,
                              hipStream_t stream)
{
    (void)in_sizes; (void)n_in; (void)out_size; (void)ws_size;
    const float* x      = (const float*)d_in[0];  // [16,384,4096]
    const float* W_bcdt = (const float*)d_in[1];  // [192,384]
    const float* W_dw   = (const float*)d_in[2];  // [192,1,3,3]
    const float* W_hz   = (const float*)d_in[3];  // [768,384]
    const float* W_out  = (const float*)d_in[4];  // [384,384]
    const float* A_p    = (const float*)d_in[5];  // [64]
    const float* D_p    = (const float*)d_in[6];  // [1]

    float* y    = (float*)d_out;                  // [16,384,4096]
    float* hout = y + 16LL * 384 * 4096;          // [16,384,64] (2nd output)

    char* w = (char*)d_ws;
    float*    bcdt = (float*)w;     w += 16LL * 192 * 4096 * 4;   // f32
    float*    dwb  = (float*)w;     w += 16LL * 192 * 4096 * 4;   // f32
    _Float16* xh   = (_Float16*)w;  w += 16LL * 384 * 4096 * 2;   // f16 copy of x
    _Float16* dwh  = (_Float16*)w;  w += 16LL * 64  * 4096 * 2;   // f16 Cm
    _Float16* abt  = (_Float16*)w;  w += 16LL * 4096 * 64 * 2;    // f16 AB^T
    float*    hbuf = (float*)w;     w += 16LL * 384 * 64 * 4;
    float*    hzb  = (float*)w;     w += 16LL * 768 * 64 * 4;
    float*    hg   = (float*)w;

    const dim3 blk(128);

    // x -> f16 (feeds G1's B and G2's A)
    cvt_f32_f16<<<dim3((16 * 384 * 4096 / 4 + 255) / 256), dim3(256), 0, stream>>>(
        x, xh, 16LL * 384 * 4096 / 4);

    // G1: bcdt[b] = W_bcdt(192x384) * xh[b](384x4096)
    gemm_wmma<float, _Float16><<<dim3(4096 / 64, 192 / 64, 16), blk, 0, stream>>>(
        W_bcdt, 0, 384, xh, 384LL * 4096, 4096,
        bcdt, 192LL * 4096, 4096, 384);

    // depthwise 3x3 SAME conv (+ f16 Cm channels)
    dwconv3x3<<<dim3(16 * 192), dim3(256), 0, stream>>>(bcdt, W_dw, dwb, dwh);

    // softmax over L + AB, store transposed f16 [l][s]
    softmax_ab<<<dim3(16 * 64), dim3(256), 0, stream>>>(dwb, A_p, abt);

    // G2: h[b] = xh[b](384x4096) * ABt[b](4096x64)
    gemm_wmma<_Float16, _Float16><<<dim3(1, 384 / 64, 16), blk, 0, stream>>>(
        xh, 384LL * 4096, 4096, abt, 4096LL * 64, 64,
        hbuf, 384LL * 64, 64, 4096);

    // G3: hz[b] = W_hz(768x384) * h[b](384x64)
    gemm_wmma<float, float><<<dim3(1, 768 / 64, 16), blk, 0, stream>>>(
        W_hz, 0, 384, hbuf, 384LL * 64, 64,
        hzb, 768LL * 64, 64, 384);

    // gate
    gate_silu<<<dim3((16 * 384 * 64 + 255) / 256), dim3(256), 0, stream>>>(hzb, D_p, hg);

    // G4: hout[b] = W_out(384x384) * hgate[b](384x64) -> second output, in place
    gemm_wmma<float, float><<<dim3(1, 384 / 64, 16), blk, 0, stream>>>(
        W_out, 0, 384, hg, 384LL * 64, 64,
        hout, 384LL * 64, 64, 384);

    // G5: y[b] = hout[b](384x64) * Cm_h[b](64x4096)
    gemm_wmma<float, _Float16><<<dim3(4096 / 64, 384 / 64, 16), blk, 0, stream>>>(
        hout, 384LL * 64, 64, dwh, 64LL * 4096, 4096,
        y, 384LL * 4096, 4096, 64);
}
